// GNNActor_1752346657363
// MI455X (gfx1250) — matrix operand order repaired and checked by reading, not compiled, not run on hardware
//
#include <hip/hip_runtime.h>
#include <hip/hip_bf16.h>
#include <math.h>

typedef __attribute__((ext_vector_type(16))) _Float16 v16h;
typedef __attribute__((ext_vector_type(8)))  float    v8f;

#define BATCH    2048
#define BPW      2            // batches per workgroup (B-fragment reuse factor)
#define ACT      14
#define CDIM     128
#define HDIM     256
#define NEDGE    182
#define MTILES   12           // ceil(182/16): 192 padded edge rows
#define NWAVES   12           // one M-tile per wave (per batch)
#define NTHREADS (NWAVES*32)  // 384 threads, wave32
#define XSTR     136          // x_s stride (f16): 272B, 16B aligned, conflict-free
#define HSTR     264          // h1/w_s stride (f16): 528B, 16B aligned, conflict-free

union AFrag { v16h h; unsigned int u[8]; };

__device__ __forceinline__ float softplus_f(float x) {
    return (x > 20.f) ? x : log1pf(expf(x));
}

// one-shot fp32 -> f16 conversion of the two 256x256 weight matrices into d_ws
__global__ void convert_weights(const float* __restrict__ w1,
                                const float* __restrict__ w2,
                                _Float16* __restrict__ w1h,
                                _Float16* __restrict__ w2h, int nel) {
    int i = blockIdx.x * blockDim.x + threadIdx.x;
    if (i < nel) {
        w1h[i] = (_Float16)w1[i];
        w2h[i] = (_Float16)w2[i];
    }
}

__global__ __launch_bounds__(NTHREADS)
void gnn_actor_kernel(const float* __restrict__ state,
                      const float* __restrict__ conv_w,
                      const float* __restrict__ conv_b,
                      const _Float16* __restrict__ w1h,
                      const float* __restrict__ lin1_b,
                      const _Float16* __restrict__ w2h,
                      const float* __restrict__ lin2_b,
                      const float* __restrict__ mu_w,
                      const float* __restrict__ mu_b,
                      const float* __restrict__ sig_w,
                      const float* __restrict__ sig_b,
                      const float* __restrict__ noise,
                      float* __restrict__ out_action,
                      float* __restrict__ out_logp)
{
    __shared__ _Float16 w_s[128 * HSTR];                 // staged half weights (66 KB)
    __shared__ _Float16 h1_s[NWAVES * BPW * 16 * HSTR];  // per-(wave,batch) h1 (198 KB)
    __shared__ _Float16 x_s[BPW * ACT * XSTR];           // residual node features
    __shared__ __align__(16) float sm[BPW * CDIM];
    __shared__ float    lp_s[BPW];

    const int tid  = threadIdx.x;
    const int b0   = blockIdx.x * BPW;
    const int wave = tid >> 5;
    const int lane = tid & 31;
    const int n    = lane & 15;   // column within 16-wide tile
    const int half = lane >> 4;   // lane-half (K / M offset role)

    if (tid < BPW) lp_s[tid] = 0.f;

    // ---------------- Phase A: fused GCN conv (both batches) ----------------
    // Fully-connected graph + self loops => deg==14, all norms 1/14:
    // conv output = relu(mean_i(state) @ conv_w^T + conv_b), identical per node.
    {
        const int bb = tid >> 7;       // 0..2; only bb<2 active
        const int c  = tid & (CDIM - 1);
        float sv[ACT];
        if (tid < BPW * CDIM) {
            const float* st = state + (size_t)(b0 + bb) * ACT * CDIM;
            float s = 0.f;
            #pragma unroll
            for (int i = 0; i < ACT; ++i) { sv[i] = st[i * CDIM + c]; s += sv[i]; }
            sm[bb * CDIM + c] = s * (1.0f / (float)ACT);
        }
        __syncthreads();
        if (tid < BPW * CDIM) {
            const float4* wr4 = (const float4*)(conv_w + (size_t)c * CDIM);
            const float4* sm4 = (const float4*)(sm + bb * CDIM);
            float m = 0.f;
            #pragma unroll 4
            for (int k = 0; k < CDIM / 4; ++k) {
                float4 w4 = wr4[k]; float4 s4 = sm4[k];
                m += w4.x * s4.x + w4.y * s4.y + w4.z * s4.z + w4.w * s4.w;
            }
            m += conv_b[c];
            float r = m > 0.f ? m : 0.f;        // relu
            #pragma unroll
            for (int i = 0; i < ACT; ++i)       // residual, store f16
                x_s[(bb * ACT + i) * XSTR + c] = (_Float16)(sv[i] + r);
        }
        __syncthreads();
    }

    const float mub = mu_b[0], sgb = sig_b[0];
    _Float16* hbuf0 = h1_s + (wave * BPW + 0) * 16 * HSTR;
    _Float16* hbuf1 = h1_s + (wave * BPW + 1) * 16 * HSTR;
    const int  mt   = wave;                      // this wave's M-tile (both batches)

    // analytic edge endpoints for A-row r = n of this M-tile
    int e = mt * 16 + n;
    int src = 0, dst = 0;
    if (e < NEDGE) {
        int i = e / 13, jj = e - i * 13;
        src = i; dst = jj + (jj >= i ? 1 : 0);
    }

    // ---- Layer-1 A fragments for both batches: ef row = concat(x[src], x[dst])
    AFrag a0[8], a1[8];
    #pragma unroll
    for (int kk = 0; kk < 8; ++kk) {
        #pragma unroll
        for (int v = 0; v < 8; ++v) {
            int kg   = kk * 32 + 2 * v + ((v >= 4) ? 8 : 0) + 8 * half;
            int node = (kg < CDIM) ? src : dst;
            int col  = kg & (CDIM - 1);
            a0[kk].u[v] = *(const unsigned int*)&x_s[node * XSTR + col];
            a1[kk].u[v] = *(const unsigned int*)&x_s[(ACT + node) * XSTR + col];
        }
    }

    // ---------------- Layer 1: h1 = leaky_relu(ef @ W1^T + b1) ----------------
    // Weights staged into shared LDS in two 128-row halves, reused by all waves;
    // each B-fragment load feeds TWO WMMAs (one per batch).
    for (int nh = 0; nh < 2; ++nh) {
        __syncthreads();
        for (int idx = tid; idx < 128 * 16; idx += NTHREADS) {
            int row = idx >> 4, ch = idx & 15;
            *(uint4*)&w_s[row * HSTR + ch * 16] =
                *(const uint4*)&w1h[(size_t)(nh * 128 + row) * HDIM + ch * 16];
        }
        __syncthreads();
        for (int ntl = 0; ntl < 8; ++ntl) {
            int nt = nh * 8 + ntl;
            const _Float16* wr = w_s + (ntl * 16 + n) * HSTR;
            v8f acc0 = {0.f,0.f,0.f,0.f,0.f,0.f,0.f,0.f};
            v8f acc1 = {0.f,0.f,0.f,0.f,0.f,0.f,0.f,0.f};
            #pragma unroll
            for (int kk = 0; kk < 8; ++kk) {
                AFrag bfr;
                #pragma unroll
                for (int v = 0; v < 8; ++v)
                    bfr.u[v] = *(const unsigned int*)&wr[kk * 32 + 2 * v + ((v >= 4) ? 8 : 0) + 8 * half];
                acc0 = __builtin_amdgcn_wmma_f32_16x16x32_f16(
                           false, a0[kk].h, false, bfr.h, (short)0, acc0, false, false);
                acc1 = __builtin_amdgcn_wmma_f32_16x16x32_f16(
                           false, a1[kk].h, false, bfr.h, (short)0, acc1, false, false);
            }
            float bias = lin1_b[nt * 16 + n];
            #pragma unroll
            for (int v = 0; v < 8; ++v) {
                float t0 = acc0[v] + bias;
                float t1 = acc1[v] + bias;
                t0 = (t0 > 0.f) ? t0 : 0.01f * t0;  // leaky_relu
                t1 = (t1 > 0.f) ? t1 : 0.01f * t1;
                hbuf0[(v + 8 * half) * HSTR + nt * 16 + n] = (_Float16)t0;
                hbuf1[(v + 8 * half) * HSTR + nt * 16 + n] = (_Float16)t1;
            }
        }
    }

    // ---- Layer-2 A fragments from this wave's own h1 tiles (DScnt keeps order)
    #pragma unroll
    for (int kk = 0; kk < 8; ++kk) {
        #pragma unroll
        for (int v = 0; v < 8; ++v) {
            int kg = kk * 32 + 2 * v + ((v >= 4) ? 8 : 0) + 8 * half;
            a0[kk].u[v] = *(const unsigned int*)&hbuf0[n * HSTR + kg];
            a1[kk].u[v] = *(const unsigned int*)&hbuf1[n * HSTR + kg];
        }
    }

    // ------- Layer 2 with fused mu/sig dot-product epilogue (h2 never stored) -------
    float mua0[8] = {0,0,0,0,0,0,0,0}, sga0[8] = {0,0,0,0,0,0,0,0};
    float mua1[8] = {0,0,0,0,0,0,0,0}, sga1[8] = {0,0,0,0,0,0,0,0};
    for (int nh = 0; nh < 2; ++nh) {
        __syncthreads();
        for (int idx = tid; idx < 128 * 16; idx += NTHREADS) {
            int row = idx >> 4, ch = idx & 15;
            *(uint4*)&w_s[row * HSTR + ch * 16] =
                *(const uint4*)&w2h[(size_t)(nh * 128 + row) * HDIM + ch * 16];
        }
        __syncthreads();
        for (int ntl = 0; ntl < 8; ++ntl) {
            int nt = nh * 8 + ntl;
            const _Float16* wr = w_s + (ntl * 16 + n) * HSTR;
            v8f acc0 = {0.f,0.f,0.f,0.f,0.f,0.f,0.f,0.f};
            v8f acc1 = {0.f,0.f,0.f,0.f,0.f,0.f,0.f,0.f};
            #pragma unroll
            for (int kk = 0; kk < 8; ++kk) {
                AFrag bfr;
                #pragma unroll
                for (int v = 0; v < 8; ++v)
                    bfr.u[v] = *(const unsigned int*)&wr[kk * 32 + 2 * v + ((v >= 4) ? 8 : 0) + 8 * half];
                acc0 = __builtin_amdgcn_wmma_f32_16x16x32_f16(
                           false, a0[kk].h, false, bfr.h, (short)0, acc0, false, false);
                acc1 = __builtin_amdgcn_wmma_f32_16x16x32_f16(
                           false, a1[kk].h, false, bfr.h, (short)0, acc1, false, false);
            }
            float bias = lin2_b[nt * 16 + n];
            float wmu  = mu_w[nt * 16 + n];
            float wsg  = sig_w[nt * 16 + n];
            #pragma unroll
            for (int v = 0; v < 8; ++v) {
                float t0 = acc0[v] + bias;
                float t1 = acc1[v] + bias;
                t0 = (t0 > 0.f) ? t0 : 0.01f * t0;  // leaky_relu
                t1 = (t1 > 0.f) ? t1 : 0.01f * t1;
                mua0[v] += t0 * wmu;  sga0[v] += t0 * wsg;
                mua1[v] += t1 * wmu;  sga1[v] += t1 * wsg;
            }
        }
    }

    // reduce the 16-lane partial dot products within each lane-half
    #pragma unroll
    for (int v = 0; v < 8; ++v) {
        #pragma unroll
        for (int m = 1; m < 16; m <<= 1) {
            mua0[v] += __shfl_xor(mua0[v], m, 32);
            sga0[v] += __shfl_xor(sga0[v], m, 32);
            mua1[v] += __shfl_xor(mua1[v], m, 32);
            sga1[v] += __shfl_xor(sga1[v], m, 32);
        }
    }

    if (n == 0) {  // lanes 0 and 16 own rows {v + 8*half}
        float lp0 = 0.f, lp1 = 0.f;
        #pragma unroll
        for (int v = 0; v < 8; ++v) {
            int e2 = mt * 16 + v + 8 * half;
            if (e2 < NEDGE) {
                float mu0 = softplus_f(mua0[v] + mub);
                float sg0 = softplus_f(sga0[v] + sgb);
                float nz0 = noise[(size_t)b0 * NEDGE + e2];
                out_action[(size_t)b0 * NEDGE + e2] = mu0 + sg0 * nz0;
                lp0 += -0.5f * nz0 * nz0 - logf(sg0) - 0.91893853320467274f;

                float mu1 = softplus_f(mua1[v] + mub);
                float sg1 = softplus_f(sga1[v] + sgb);
                float nz1 = noise[(size_t)(b0 + 1) * NEDGE + e2];
                out_action[(size_t)(b0 + 1) * NEDGE + e2] = mu1 + sg1 * nz1;
                lp1 += -0.5f * nz1 * nz1 - logf(sg1) - 0.91893853320467274f;
            }
        }
        atomicAdd(&lp_s[0], lp0);   // ds_add_f32
        atomicAdd(&lp_s[1], lp1);
    }
    __syncthreads();
    if (tid < BPW) out_logp[b0 + tid] = lp_s[tid];
}

extern "C" void kernel_launch(void* const* d_in, const int* in_sizes, int n_in,
                              void* d_out, int out_size, void* d_ws, size_t ws_size,
                              hipStream_t stream)
{
    (void)in_sizes; (void)n_in; (void)out_size; (void)ws_size;
    const float* state  = (const float*)d_in[0];
    const float* conv_w = (const float*)d_in[1];
    const float* conv_b = (const float*)d_in[2];
    const float* lin1_w = (const float*)d_in[3];
    const float* lin1_b = (const float*)d_in[4];
    const float* lin2_w = (const float*)d_in[5];
    const float* lin2_b = (const float*)d_in[6];
    const float* mu_w   = (const float*)d_in[7];
    const float* mu_b   = (const float*)d_in[8];
    const float* sig_w  = (const float*)d_in[9];
    const float* sig_b  = (const float*)d_in[10];
    const float* noise  = (const float*)d_in[11];
    // d_in[12]/d_in[13] (edge_index/edges): the all-pairs structure is recomputed
    // analytically on device, so these are not needed.

    _Float16* w1h = (_Float16*)d_ws;                 // 256*256 f16 = 128 KB
    _Float16* w2h = w1h + HDIM * HDIM;               // next 128 KB

    float* out_action = (float*)d_out;               // (B, 182)
    float* out_logp   = out_action + (size_t)BATCH * NEDGE;  // (B,)

    convert_weights<<<(HDIM * HDIM + 255) / 256, 256, 0, stream>>>(
        lin1_w, lin2_w, w1h, w2h, HDIM * HDIM);

    gnn_actor_kernel<<<BATCH / BPW, NTHREADS, 0, stream>>>(
        state, conv_w, conv_b, w1h, lin1_b, w2h, lin2_b,
        mu_w, mu_b, sig_w, sig_b, noise, out_action, out_logp);
}
